// Qwen2MoeMLPStack_74457553043885
// MI455X (gfx1250) — compile-verified, matrix operand order checked by reading, not run on hardware
//
#include <hip/hip_runtime.h>
#include <hip/hip_bf16.h>

// ---------------------------------------------------------------------------
// MoE SwiGLU MLP (Qwen2-style) for gfx1250 (MI455X, wave32, WMMA bf16)
//   gate = x @ w_gate[e] ; up = x @ w_up[e] ; h = silu(gate)*up
//   out  = h @ w_down[e]
// fp32 -> bf16 operand conversion (native __bf16 cvt), WMMA bf16 with fp32
// accumulation, BK=64 double-pumped K steps, async LDS DMA for bf16 A tiles.
// ---------------------------------------------------------------------------

#define NUM_EXPERTS 16
#define HIDDEN      2048
#define INTER       1408
#define TOKENS      32768

#define BM 128
#define BN 64
#define BK 64
#define LDS_STRIDE 72   // bf16 elements per row (144B = 9*16B): 64 payload + 8 pad

typedef __attribute__((ext_vector_type(16))) __bf16 v16bf;
typedef __attribute__((ext_vector_type(8)))  float  v8f;

union Frag {
    v16bf v;
    uint4 q[2];
};

__device__ __forceinline__ unsigned short f2bf(float f) {
    return __builtin_bit_cast(unsigned short, (__bf16)f);   // native cvt path
}
__device__ __forceinline__ unsigned pk2bf(float lo, float hi) {
    return (unsigned)f2bf(lo) | ((unsigned)f2bf(hi) << 16);
}

__device__ __forceinline__ int tile_expert(const int* gs, int t0) {
    int e = 0, cum = 0;
    while (e < NUM_EXPERTS - 1 && t0 >= cum + gs[e]) { cum += gs[e]; ++e; }
    return e;
}

// 32-bit LDS byte offset for gfx1250 async/DS ops (generic -> addrspace(3))
__device__ __forceinline__ unsigned lds_off32(const void* p) {
    return (unsigned)(uintptr_t)(__attribute__((address_space(3))) const void*)p;
}

// Load a 16x32 bf16 fragment (WMMA A/B-transposed layout) from LDS.
// lane&15 = row, lane>=16 selects K halves {8..15,24..31} vs {0..7,16..23};
// sub selects K base (0 or 32) within the BK=64 row.
__device__ __forceinline__ Frag load_frag(const unsigned short* lds,
                                          int base_row, int lane, int sub) {
    const int lr   = lane & 15;
    const int half = lane >> 4;
    const unsigned short* p = lds + (base_row + lr) * LDS_STRIDE + sub * 32 + half * 8;
    Frag f;
    f.q[0] = *(const uint4*)(p);        // K 0..7   (or 8..15)
    f.q[1] = *(const uint4*)(p + 16);   // K 16..23 (or 24..31)
    return f;
}

// ---------------------------------------------------------------------------
// Kernel 1: gate/up grouped GEMM + fused SwiGLU -> bf16 h in workspace
// grid = (INTER/BN, TOKENS/BM), block = 256 (8 waves)
// ---------------------------------------------------------------------------
__global__ __launch_bounds__(256)
void moe_gateup_kernel(const float* __restrict__ x,
                       const float* __restrict__ w_gate,
                       const float* __restrict__ w_up,
                       const int*   __restrict__ group_sizes,
                       unsigned short* __restrict__ h_out) {
    __shared__ __align__(16) unsigned short lda [BM * LDS_STRIDE];   // 18 KB
    __shared__ __align__(16) unsigned short ldbg[BN * LDS_STRIDE];   //  9 KB
    __shared__ __align__(16) unsigned short ldbu[BN * LDS_STRIDE];   //  9 KB

    const int tid  = threadIdx.x;
    const int wave = tid >> 5;
    const int lane = tid & 31;

    const int n0 = blockIdx.x * BN;
    const int t0 = blockIdx.y * BM;

    const int e = tile_expert(group_sizes, t0);
    const size_t wbase = (size_t)e * HIDDEN * INTER;
    const float* wg = w_gate + wbase;
    const float* wu = w_up   + wbase;

    v8f accg[4], accu[4];
#pragma unroll
    for (int i = 0; i < 4; ++i) {
        accg[i] = (v8f){0.f,0.f,0.f,0.f,0.f,0.f,0.f,0.f};
        accu[i] = (v8f){0.f,0.f,0.f,0.f,0.f,0.f,0.f,0.f};
    }

    for (int k0 = 0; k0 < HIDDEN; k0 += BK) {
        // ---- stage A tile (BM x BK fp32 -> bf16, row-major) ----
#pragma unroll
        for (int it = 0; it < 8; ++it) {               // 2048 float4 chunks
            int c   = tid + 256 * it;
            int row = c >> 4;                          // BK/4 = 16 chunks/row
            int c4  = c & 15;
            const float4 f = *(const float4*)(x + (size_t)(t0 + row) * HIDDEN
                                                + k0 + c4 * 4);
            uint2 packed;
            packed.x = pk2bf(f.x, f.y);
            packed.y = pk2bf(f.z, f.w);
            *(uint2*)(&lda[row * LDS_STRIDE + c4 * 4]) = packed;
        }
        // ---- stage B tiles transposed (BK x BN fp32 -> LDS[n][k] bf16) ----
#pragma unroll
        for (int it = 0; it < 4; ++it) {               // 1024 float4 chunks
            int c  = tid + 256 * it;
            int kk = c >> 4;                           // row (K) index
            int n4 = c & 15;                           // float4 along N
            const size_t go = (size_t)(k0 + kk) * INTER + n0 + n4 * 4;
            const float4 g = *(const float4*)(wg + go);
            const float4 u = *(const float4*)(wu + go);
            const int nb = n4 * 4;
            ldbg[(nb + 0) * LDS_STRIDE + kk] = f2bf(g.x);
            ldbg[(nb + 1) * LDS_STRIDE + kk] = f2bf(g.y);
            ldbg[(nb + 2) * LDS_STRIDE + kk] = f2bf(g.z);
            ldbg[(nb + 3) * LDS_STRIDE + kk] = f2bf(g.w);
            ldbu[(nb + 0) * LDS_STRIDE + kk] = f2bf(u.x);
            ldbu[(nb + 1) * LDS_STRIDE + kk] = f2bf(u.y);
            ldbu[(nb + 2) * LDS_STRIDE + kk] = f2bf(u.z);
            ldbu[(nb + 3) * LDS_STRIDE + kk] = f2bf(u.w);
        }
        // prefetch next K-slab of weights (global_prefetch_b8, speculative)
        if (k0 + BK < HIDDEN) {
            __builtin_prefetch(wg + (size_t)(k0 + BK) * INTER + n0, 0, 0);
            __builtin_prefetch(wu + (size_t)(k0 + BK) * INTER + n0, 0, 0);
        }
        __syncthreads();

        // ---- compute: each wave owns a 16-row strip, full BN=64 ----
#pragma unroll
        for (int s = 0; s < 2; ++s) {                  // two K=32 sub-steps
            const Frag fa = load_frag(lda, wave * 16, lane, s);
            Frag fbg[4], fbu[4];
#pragma unroll
            for (int nf = 0; nf < 4; ++nf) fbg[nf] = load_frag(ldbg, nf * 16, lane, s);
#pragma unroll
            for (int nf = 0; nf < 4; ++nf) fbu[nf] = load_frag(ldbu, nf * 16, lane, s);
#pragma unroll
            for (int nf = 0; nf < 4; ++nf)
                accg[nf] = __builtin_amdgcn_wmma_f32_16x16x32_bf16(
                    false, fa.v, false, fbg[nf].v, (short)0, accg[nf], false, false);
#pragma unroll
            for (int nf = 0; nf < 4; ++nf)
                accu[nf] = __builtin_amdgcn_wmma_f32_16x16x32_bf16(
                    false, fa.v, false, fbu[nf].v, (short)0, accu[nf], false, false);
        }
        __syncthreads();
    }

    // ---- epilogue: h = silu(gate) * up, bf16 store to workspace ----
    const int lr   = lane & 15;
    const int half = lane >> 4;
#pragma unroll
    for (int nf = 0; nf < 4; ++nf) {
        const int col = n0 + nf * 16 + lr;
#pragma unroll
        for (int j = 0; j < 8; ++j) {
            const int row = t0 + wave * 16 + j + half * 8;
            const float g = accg[nf][j];
            const float u = accu[nf][j];
            const float s = g / (1.0f + __expf(-g));
            h_out[(size_t)row * INTER + col] = f2bf(s * u);
        }
    }
}

// ---------------------------------------------------------------------------
// Kernel 2: down-projection grouped GEMM, bf16 h @ w_down -> fp32 out
// A tile staged with GLOBAL_LOAD_ASYNC_TO_LDS_B128 (ASYNCcnt DMA path).
// grid = (HIDDEN/BN, TOKENS/BM), block = 256 (8 waves)
// ---------------------------------------------------------------------------
__global__ __launch_bounds__(256)
void moe_down_kernel(const unsigned short* __restrict__ h_in,
                     const float* __restrict__ w_down,
                     const int*   __restrict__ group_sizes,
                     float* __restrict__ out) {
    __shared__ __align__(16) unsigned short lda[BM * LDS_STRIDE];    // 18 KB
    __shared__ __align__(16) unsigned short ldb[BN * LDS_STRIDE];    //  9 KB

    const int tid  = threadIdx.x;
    const int wave = tid >> 5;
    const int lane = tid & 31;

    const int n0 = blockIdx.x * BN;
    const int t0 = blockIdx.y * BM;

    const int e = tile_expert(group_sizes, t0);
    const float* wd = w_down + (size_t)e * INTER * HIDDEN;

    v8f acc[4];
#pragma unroll
    for (int i = 0; i < 4; ++i)
        acc[i] = (v8f){0.f,0.f,0.f,0.f,0.f,0.f,0.f,0.f};

    for (int k0 = 0; k0 < INTER; k0 += BK) {
        // ---- stage A tile (BM x BK bf16) via async LDS DMA, 16B chunks ----
#pragma unroll
        for (int it = 0; it < 4; ++it) {               // 1024 uint4 chunks
            int c   = tid + 256 * it;
            int row = c >> 3;                          // BK/8 = 8 chunks/row
            int c8  = c & 7;
            const unsigned long long src =
                (unsigned long long)(h_in + (size_t)(t0 + row) * INTER + k0 + c8 * 8);
            const unsigned dst = lds_off32(&lda[row * LDS_STRIDE + c8 * 8]);
            asm volatile("global_load_async_to_lds_b128 %0, %1, off"
                         :: "v"(dst), "v"(src) : "memory");
        }
        // ---- stage B tile transposed (BK x BN fp32 -> LDS[n][k] bf16) ----
#pragma unroll
        for (int it = 0; it < 4; ++it) {               // 1024 float4 chunks
            int c  = tid + 256 * it;
            int kk = c >> 4;
            int n4 = c & 15;
            const float4 w = *(const float4*)(wd + (size_t)(k0 + kk) * HIDDEN
                                                 + n0 + n4 * 4);
            const int nb = n4 * 4;
            ldb[(nb + 0) * LDS_STRIDE + kk] = f2bf(w.x);
            ldb[(nb + 1) * LDS_STRIDE + kk] = f2bf(w.y);
            ldb[(nb + 2) * LDS_STRIDE + kk] = f2bf(w.z);
            ldb[(nb + 3) * LDS_STRIDE + kk] = f2bf(w.w);
        }
        if (k0 + BK < INTER) {
            __builtin_prefetch(wd + (size_t)(k0 + BK) * HIDDEN + n0, 0, 0);
        }
        asm volatile("s_wait_asynccnt 0" ::: "memory");   // A-tile DMA done
        __syncthreads();

#pragma unroll
        for (int s = 0; s < 2; ++s) {
            const Frag fa = load_frag(lda, wave * 16, lane, s);
            Frag fb[4];
#pragma unroll
            for (int nf = 0; nf < 4; ++nf) fb[nf] = load_frag(ldb, nf * 16, lane, s);
#pragma unroll
            for (int nf = 0; nf < 4; ++nf)
                acc[nf] = __builtin_amdgcn_wmma_f32_16x16x32_bf16(
                    false, fa.v, false, fb[nf].v, (short)0, acc[nf], false, false);
        }
        __syncthreads();
    }

    // ---- epilogue: fp32 store (coalesced across lanes) ----
    const int lr   = lane & 15;
    const int half = lane >> 4;
#pragma unroll
    for (int nf = 0; nf < 4; ++nf) {
        const int col = n0 + nf * 16 + lr;
#pragma unroll
        for (int j = 0; j < 8; ++j) {
            const int row = t0 + wave * 16 + j + half * 8;
            out[(size_t)row * HIDDEN + col] = acc[nf][j];
        }
    }
}

// ---------------------------------------------------------------------------
extern "C" void kernel_launch(void* const* d_in, const int* in_sizes, int n_in,
                              void* d_out, int out_size, void* d_ws, size_t ws_size,
                              hipStream_t stream) {
    const float* x      = (const float*)d_in[0];
    const float* w_gate = (const float*)d_in[1];
    const float* w_up   = (const float*)d_in[2];
    const float* w_down = (const float*)d_in[3];
    const int*   gs     = (const int*)d_in[4];
    float* out = (float*)d_out;

    // bf16 intermediate h [TOKENS x INTER] in workspace (92 MB)
    unsigned short* h = (unsigned short*)d_ws;

    dim3 block(256);
    dim3 grid1(INTER / BN,  TOKENS / BM);   // (22, 256)
    dim3 grid2(HIDDEN / BN, TOKENS / BM);   // (32, 256)

    moe_gateup_kernel<<<grid1, block, 0, stream>>>(x, w_gate, w_up, gs, h);
    moe_down_kernel  <<<grid2, block, 0, stream>>>(h, w_down, gs, out);
}
